// MoDLlamaDecoderLayer_55207509623073
// MI455X (gfx1250) — compile-verified, matrix operand order checked by reading, not run on hardware
//
#include <hip/hip_runtime.h>
#include <hip/hip_bf16.h>

// ---------------------------------------------------------------------------
// MoD Llama decoder layer for MI455X (gfx1250, wave32).
// Heavy math: v_wmma_f32_16x16x32_bf16 (f32 accumulate).
// GEMM tile staging: Tensor Data Mover (tensor_load_to_lds), double-buffered,
// synchronized with s_wait_tensorcnt; falls back to manual LDS staging if the
// TDM builtin is unavailable.
// ---------------------------------------------------------------------------

typedef __attribute__((ext_vector_type(16))) __bf16 v16bf;
typedef __attribute__((ext_vector_type(8)))  float  v8f;

#define WMMA_BF16(a, b, c) \
  __builtin_amdgcn_wmma_f32_16x16x32_bf16(false, (a), false, (b), (short)0, (c), false, false)

constexpr int Bc = 2, Sc = 2048, Dc = 2048, Hc = 16, HDc = 128, Fc = 8192;
constexpr float EPSc = 1e-5f;

static __device__ __forceinline__ int imin(int a, int b) { return a < b ? a : b; }

// ---------------------------------------------------------------------------
// Tensor Data Mover: 2-D bf16 tile (rows x 32 elems) global -> LDS.
// D# per CDNA5 ISA. pad_enable inserts 32B of LDS padding
// after every 64B row chunk -> LDS row stride 96B (48 bf16), which is exactly
// the bank-conflict padding the WMMA fragment loads use.
// ---------------------------------------------------------------------------
#if __has_builtin(__builtin_amdgcn_tensor_load_to_lds)
#define HAVE_TDM 1
typedef unsigned int u32x4 __attribute__((ext_vector_type(4)));
typedef int i32x4 __attribute__((ext_vector_type(4)));
typedef int i32x8 __attribute__((ext_vector_type(8)));

#define LDSA(p) ((unsigned)(unsigned long long)(const void*)(p))

static __device__ __forceinline__ void tdm_load_tile_bf16(unsigned lds_byte_addr,
                                                          const __bf16* gptr, int rows,
                                                          long long stride_elems) {
  const unsigned long long ga = (unsigned long long)(const void*)gptr;
  u32x4 g0;
  g0[0] = 1u;                                   // count=1 (valid user descriptor)
  g0[1] = lds_byte_addr;                        // lds_addr [63:32]
  g0[2] = (unsigned)(ga & 0xffffffffu);         // global_addr [95:64]
  g0[3] = (unsigned)((ga >> 32) & 0x01ffffffu)  // global_addr [120:96]
          | (2u << 30);                         // type = 2 ("image")
  const unsigned td0 = 32u, td1 = (unsigned)rows;      // tensor dims (elems)
  const unsigned tl0 = 32u, tl1 = (unsigned)rows;      // tile dims (elems)
  const unsigned long long st0 = (unsigned long long)stride_elems;
  i32x8 g1;
  g1[0] = (int)((1u << 16)      // data_size = 1 -> 2 bytes
                | (1u << 20)    // pad_enable
                | (3u << 22)    // pad_interval: every 16 DWORDs (64B)
                | (7u << 25));  // pad_amount: 8 DWORDs (32B)
  g1[1] = (int)((td0 & 0xffffu) << 16);                                   // tensor_dim0 lo
  g1[2] = (int)(((td0 >> 16) & 0xffffu) | ((td1 & 0xffffu) << 16));       // dim0 hi | dim1 lo
  g1[3] = (int)(((td1 >> 16) & 0xffffu) | (tl0 << 16));                   // dim1 hi | tile_dim0
  g1[4] = (int)tl1;                                                       // tile_dim1 (tile_dim2=0)
  g1[5] = (int)(st0 & 0xffffffffu);                                       // dim0_stride lo
  g1[6] = (int)((st0 >> 32) & 0xffffu);                                   // dim0_stride hi
  g1[7] = 0;
  const i32x4 gz = {0, 0, 0, 0};
#if __clang_major__ >= 23
  const i32x8 gz8 = {0, 0, 0, 0, 0, 0, 0, 0};
  __builtin_amdgcn_tensor_load_to_lds(g0, g1, gz, gz, gz8, 0);
#else
  __builtin_amdgcn_tensor_load_to_lds(g0, g1, gz, gz, 0);
#endif
}
#endif  // HAVE_TDM

// ---------------------------------------------------------------------------
// f32 -> bf16 conversion (weights, one-time)
// ---------------------------------------------------------------------------
__global__ void f32_to_bf16_kernel(const float* __restrict__ src, __bf16* __restrict__ dst,
                                   long long n) {
  long long i = ((long long)blockIdx.x * blockDim.x + threadIdx.x) * 4;
  if (i + 3 < n) {
    float4 v4 = *(const float4*)(src + i);
    dst[i + 0] = (__bf16)v4.x;
    dst[i + 1] = (__bf16)v4.y;
    dst[i + 2] = (__bf16)v4.z;
    dst[i + 3] = (__bf16)v4.w;
  } else {
    for (; i < n; i++) dst[i] = (__bf16)src[i];
  }
}

// ---------------------------------------------------------------------------
// sum(v_mask) -> top_k on device (no host sync; graph-capture safe)
// ---------------------------------------------------------------------------
__global__ void vmask_topk_kernel(const float* __restrict__ vmask, int n, int* __restrict__ kdev) {
  __shared__ float red[256];
  float s = 0.f;
  for (int i = threadIdx.x; i < n; i += 256) s += vmask[i];
  red[threadIdx.x] = s;
  __syncthreads();
  for (int st = 128; st > 0; st >>= 1) {
    if (threadIdx.x < st) red[threadIdx.x] += red[threadIdx.x + st];
    __syncthreads();
  }
  if (threadIdx.x == 0) {
    float fl = red[0];
    float cap = 1.f - fl * 0.5f / (float)Sc;
    int tk = (int)ceilf((float)Sc * cap);
    if (tk < 1) tk = 1;
    if (tk > Sc) tk = Sc;
    kdev[0] = tk;
  }
}

// ---------------------------------------------------------------------------
// Fused RMSNorm + router logits + log-softmax, one block per token
// ---------------------------------------------------------------------------
__global__ void router_kernel(const float* __restrict__ hidden, const float* __restrict__ ln1w,
                              const float* __restrict__ rw, const float* __restrict__ rb,
                              float* __restrict__ route_prob, float* __restrict__ logp) {
  const int s = blockIdx.x, b = blockIdx.y, tid = threadIdx.x;
  const float* x = hidden + ((long long)b * Sc + s) * Dc;
  float ss = 0.f, t0 = 0.f, t1 = 0.f;
  for (int c = tid; c < Dc; c += 256) {
    float xv = x[c];
    ss += xv * xv;
    float xw = xv * ln1w[c];
    t0 += xw * rw[c];
    t1 += xw * rw[Dc + c];
  }
  __shared__ float r0[256], r1[256], r2[256];
  r0[tid] = ss; r1[tid] = t0; r2[tid] = t1;
  __syncthreads();
  for (int st = 128; st > 0; st >>= 1) {
    if (tid < st) { r0[tid] += r0[tid + st]; r1[tid] += r1[tid + st]; r2[tid] += r2[tid + st]; }
    __syncthreads();
  }
  if (tid == 0) {
    float rms = rsqrtf(r0[0] / (float)Dc + EPSc);
    float l0 = r1[0] * rms + rb[0];
    float l1 = r2[0] * rms + rb[1];
    float m = fmaxf(l0, l1);
    float lse = m + logf(__expf(l0 - m) + __expf(l1 - m));
    route_prob[b * Sc + s] = __expf(l1 - lse);
    logp[(b * Sc + s) * 2 + 0] = l0 - lse;
    logp[(b * Sc + s) * 2 + 1] = l1 - lse;
  }
}

// ---------------------------------------------------------------------------
// Exact top-k selection by rank counting (jax top_k tie-break: lower index wins)
// ---------------------------------------------------------------------------
__global__ void rank_kernel(const float* __restrict__ route_prob, const float* __restrict__ vmask,
                            const int* __restrict__ kdev, int* __restrict__ selflag) {
  const int b = blockIdx.x, tid = threadIdx.x;
  __shared__ float pp[Sc];
  for (int i = tid; i < Sc; i += 256)
    pp[i] = route_prob[b * Sc + i] + (1.f - vmask[b * Sc + i]);
  __syncthreads();
  const int tk = *kdev;
  for (int i = tid; i < Sc; i += 256) {
    float pi = pp[i];
    int rank = 0;
    for (int j = 0; j < Sc; j++) {
      float pj = pp[j];
      rank += (pj > pi) || ((pj == pi) && (j < i));
    }
    selflag[b * Sc + i] = (rank < tk) ? 1 : 0;
  }
}

__global__ void compact_kernel(const int* __restrict__ selflag, const float* __restrict__ route_prob,
                               const float* __restrict__ logp, int* __restrict__ sel_idx,
                               float* __restrict__ rwt, float* __restrict__ auxpart) {
  const int b = blockIdx.x;
  int pos = 0;
  float aux = 0.f;
  for (int i = 0; i < Sc; i++) {
    if (selflag[b * Sc + i]) {
      sel_idx[b * Sc + pos] = i;
      rwt[b * Sc + pos] = route_prob[b * Sc + i];
      pos++;
      aux += logp[(b * Sc + i) * 2 + 1];
    } else {
      aux += logp[(b * Sc + i) * 2 + 0];
    }
  }
  auxpart[b] = aux;
}

__global__ void aux_finalize_kernel(const float* __restrict__ auxpart, float* __restrict__ out) {
  out[(long long)Bc * Sc * Dc] = -(auxpart[0] + auxpart[1]) / (float)(Bc * Sc);
}

// ---------------------------------------------------------------------------
// RMSNorm of (optionally gathered) f32 rows -> bf16 rows
// ---------------------------------------------------------------------------
__global__ void rmsnorm_bf16_kernel(const float* __restrict__ src, const int* __restrict__ gather,
                                    const float* __restrict__ w, __bf16* __restrict__ dst,
                                    const int* __restrict__ kdev) {
  const int row = blockIdx.x, b = blockIdx.y, tid = threadIdx.x;
  if (row >= *kdev) return;
  const int srow = gather ? gather[b * Sc + row] : row;
  const float* x = src + ((long long)b * Sc + srow) * Dc;
  __bf16* o = dst + ((long long)b * Sc + row) * Dc;
  float ss = 0.f;
  for (int c = tid; c < Dc; c += 256) { float xv = x[c]; ss += xv * xv; }
  __shared__ float red[256];
  red[tid] = ss;
  __syncthreads();
  for (int st = 128; st > 0; st >>= 1) {
    if (tid < st) red[tid] += red[tid + st];
    __syncthreads();
  }
  const float rms = rsqrtf(red[0] / (float)Dc + EPSc);
  for (int c = tid; c < Dc; c += 256) o[c] = (__bf16)(x[c] * rms * w[c]);
}

// ---------------------------------------------------------------------------
// Tiled bf16 WMMA GEMM:  C[M,N] = A[M,K] * W[N,K]^T   (f32 accumulate)
// Block tile 128x128, BK=32, 256 threads = 8 waves (wave tile 32x64).
// Tiles staged via TDM (double-buffered, s_wait_tensorcnt) when available.
// MODE 0: store bf16. MODE 1: f32 store = gather(hidden) + acc (o-proj residual).
// MODE 2: scatter to d_out: hidden_out[b,tok,:] = xmid + acc * r_weight (down-proj).
// ---------------------------------------------------------------------------
template <int MODE>
__global__ __launch_bounds__(256) void gemm_bf16_wmma(
    const __bf16* __restrict__ A, long long strideA, const __bf16* __restrict__ W,
    void* __restrict__ Cout, long long strideC, int N, int K, const int* __restrict__ kdev,
    const float* __restrict__ resid, const int* __restrict__ sel_idx,
    const float* __restrict__ rweights) {
  const int Md = *kdev;
  const int m0 = blockIdx.y * 128;
  if (m0 >= Md) return;
  const int n0 = blockIdx.x * 128;
  const int bz = blockIdx.z;
  const __bf16* Ab = A + (long long)bz * strideA;

  // row stride 48 bf16 = 96B (TDM pad: 64B data + 32B pad per row)
  __shared__ __align__(32) __bf16 As[2][128 * 48];
  __shared__ __align__(32) __bf16 Ws[2][128 * 48];

  const int tid = threadIdx.x;
  const int lane = tid & 31, wv = tid >> 5;
  const int lh = lane >> 4, l16 = lane & 15;
  const int wr = (wv & 3) * 32;    // wave row base in tile
  const int wc = (wv >> 2) * 64;   // wave col base in tile

  v8f acc[2][4] = {};

#if defined(HAVE_TDM)
  const int aRows = imin(128, Md - m0);  // TDM OOB rows read as zero; never stored
  if (wv == 0) {
    tdm_load_tile_bf16(LDSA(&As[0][0]), Ab + (long long)m0 * K, aRows, K);
    tdm_load_tile_bf16(LDSA(&Ws[0][0]), W + (long long)n0 * K, 128, K);
  }
  for (int k0 = 0; k0 < K; k0 += 32) {
    const int buf = (k0 >> 5) & 1;
    const bool more = (k0 + 32) < K;
    if (wv == 0) {
      if (more) {  // prefetch next tiles into the other buffer, then wait for current
        tdm_load_tile_bf16(LDSA(&As[buf ^ 1][0]), Ab + (long long)m0 * K + (k0 + 32), aRows, K);
        tdm_load_tile_bf16(LDSA(&Ws[buf ^ 1][0]), W + (long long)n0 * K + (k0 + 32), 128, K);
        __builtin_amdgcn_s_wait_tensorcnt(2);
      } else {
        __builtin_amdgcn_s_wait_tensorcnt(0);
      }
    }
    __syncthreads();
    v16bf a0 = *(const v16bf*)&As[buf][(wr + l16) * 48 + lh * 16];
    v16bf a1 = *(const v16bf*)&As[buf][(wr + 16 + l16) * 48 + lh * 16];
    v16bf b0 = *(const v16bf*)&Ws[buf][(wc + l16) * 48 + lh * 16];
    v16bf b1 = *(const v16bf*)&Ws[buf][(wc + 16 + l16) * 48 + lh * 16];
    v16bf b2 = *(const v16bf*)&Ws[buf][(wc + 32 + l16) * 48 + lh * 16];
    v16bf b3 = *(const v16bf*)&Ws[buf][(wc + 48 + l16) * 48 + lh * 16];
    acc[0][0] = WMMA_BF16(a0, b0, acc[0][0]);
    acc[0][1] = WMMA_BF16(a0, b1, acc[0][1]);
    acc[0][2] = WMMA_BF16(a0, b2, acc[0][2]);
    acc[0][3] = WMMA_BF16(a0, b3, acc[0][3]);
    acc[1][0] = WMMA_BF16(a1, b0, acc[1][0]);
    acc[1][1] = WMMA_BF16(a1, b1, acc[1][1]);
    acc[1][2] = WMMA_BF16(a1, b2, acc[1][2]);
    acc[1][3] = WMMA_BF16(a1, b3, acc[1][3]);
    __syncthreads();
  }
#else
  // Fallback: cooperative vector staging
  const int ldRow = tid >> 1;
  const int ldCol = (tid & 1) * 16;
  const int aRow = imin(m0 + ldRow, Md - 1);
  const __bf16* aPtr = Ab + (long long)aRow * K + ldCol;
  const __bf16* wPtr = W + (long long)(n0 + ldRow) * K + ldCol;
  for (int k0 = 0; k0 < K; k0 += 32) {
    *(v16bf*)&As[0][ldRow * 48 + ldCol] = *(const v16bf*)(aPtr + k0);
    *(v16bf*)&Ws[0][ldRow * 48 + ldCol] = *(const v16bf*)(wPtr + k0);
    __builtin_prefetch(aPtr + k0 + 128, 0, 1);
    __builtin_prefetch(wPtr + k0 + 128, 0, 1);
    __syncthreads();
    v16bf a0 = *(const v16bf*)&As[0][(wr + l16) * 48 + lh * 16];
    v16bf a1 = *(const v16bf*)&As[0][(wr + 16 + l16) * 48 + lh * 16];
    v16bf b0 = *(const v16bf*)&Ws[0][(wc + l16) * 48 + lh * 16];
    v16bf b1 = *(const v16bf*)&Ws[0][(wc + 16 + l16) * 48 + lh * 16];
    v16bf b2 = *(const v16bf*)&Ws[0][(wc + 32 + l16) * 48 + lh * 16];
    v16bf b3 = *(const v16bf*)&Ws[0][(wc + 48 + l16) * 48 + lh * 16];
    acc[0][0] = WMMA_BF16(a0, b0, acc[0][0]);
    acc[0][1] = WMMA_BF16(a0, b1, acc[0][1]);
    acc[0][2] = WMMA_BF16(a0, b2, acc[0][2]);
    acc[0][3] = WMMA_BF16(a0, b3, acc[0][3]);
    acc[1][0] = WMMA_BF16(a1, b0, acc[1][0]);
    acc[1][1] = WMMA_BF16(a1, b1, acc[1][1]);
    acc[1][2] = WMMA_BF16(a1, b2, acc[1][2]);
    acc[1][3] = WMMA_BF16(a1, b3, acc[1][3]);
    __syncthreads();
  }
#endif

#pragma unroll
  for (int r = 0; r < 2; r++)
#pragma unroll
    for (int c = 0; c < 4; c++)
#pragma unroll
      for (int i = 0; i < 8; i++) {
        // C layout: VGPR i -> row = i + 8*(lane/16); col = lane%16
        const int row = m0 + wr + r * 16 + lh * 8 + i;
        const int col = n0 + wc + c * 16 + l16;
        if (row >= Md) continue;
        const float v = acc[r][c][i];
        if (MODE == 0) {
          ((__bf16*)Cout)[(long long)bz * strideC + (long long)row * N + col] = (__bf16)v;
        } else if (MODE == 1) {
          const int tok = sel_idx[bz * Sc + row];
          ((float*)Cout)[(long long)bz * strideC + (long long)row * N + col] =
              resid[((long long)bz * Sc + tok) * Dc + col] + v;
        } else {
          const int tok = sel_idx[bz * Sc + row];
          const float outv =
              resid[((long long)bz * Sc + row) * Dc + col] + v * rweights[bz * Sc + row];
          ((float*)Cout)[((long long)bz * Sc + tok) * (long long)Dc + col] = outv;
        }
      }
}

// ---------------------------------------------------------------------------
// RoPE on q (with 1/sqrt(HD) folded in) and k, bf16 in-place, fresh positions.
// ---------------------------------------------------------------------------
__global__ void rope_kernel(__bf16* __restrict__ q, __bf16* __restrict__ k,
                            const int* __restrict__ kdev) {
  const int row = blockIdx.x, b = blockIdx.y;
  if (row >= *kdev) return;
  const long long base = ((long long)b * Sc + row) * Dc;
  const float lb = logf(10000.f);
  const float qs = 0.08838834764831845f;  // 1/sqrt(128)
  for (int p = threadIdx.x; p < Dc / 2; p += 256) {
    const int h = p >> 6, d = p & 63;
    const float inv = __expf(-((2.f * (float)d) / (float)HDc) * lb);
    const float ang = (float)row * inv;
    const float cs = __cosf(ang), sn = __sinf(ang);
    const long long i0 = base + h * HDc + d, i1 = i0 + 64;
    const float q1 = (float)q[i0], q2 = (float)q[i1];
    q[i0] = (__bf16)((q1 * cs - q2 * sn) * qs);
    q[i1] = (__bf16)((q2 * cs + q1 * sn) * qs);
    const float k1 = (float)k[i0], k2 = (float)k[i1];
    k[i0] = (__bf16)(k1 * cs - k2 * sn);
    k[i1] = (__bf16)(k2 * cs + k1 * sn);
  }
}

// ---------------------------------------------------------------------------
// Flash attention, causal, 64-query tile x 64-key tiles, WMMA for QK^T and PV.
// Block = 128 threads (4 waves); wave owns 16 query rows. K and V tiles are
// staged in LDS so WMMA operands come from ds_load (no per-WMMA global waits).
// ---------------------------------------------------------------------------
__global__ __launch_bounds__(128) void flash_attn_kernel(
    const __bf16* __restrict__ q, const __bf16* __restrict__ k, const __bf16* __restrict__ v,
    __bf16* __restrict__ out, const int* __restrict__ kdev) {
  const int Md = *kdev;
  const int qt = blockIdx.x;
  if (qt * 64 >= Md) return;
  const int h = blockIdx.y, b = blockIdx.z;
  const int tid = threadIdx.x, lane = tid & 31, wv = tid >> 5;
  const int lh = lane >> 4, l16 = lane & 15;
  const long long bOff = (long long)b * Sc * Dc;
  const int hOff = h * HDc;
  const int qbase = qt * 64 + wv * 16;

  // Q fragments (A-matrix 16x32 per k-step), loaded once from global
  v16bf qf[4];
  {
    const int row = imin(qbase + l16, Md - 1);
    const __bf16* qp = q + bOff + (long long)row * Dc + hOff + lh * 16;
#pragma unroll
    for (int kk = 0; kk < 4; kk++) qf[kk] = *(const v16bf*)(qp + kk * 32);
  }

  v8f oacc[8] = {};
  float mrow[8], lrow[8];
#pragma unroll
  for (int i = 0; i < 8; i++) { mrow[i] = -1e30f; lrow[i] = 0.f; }

  __shared__ __align__(32) __bf16 Ks[64 * 144];  // [key][hd],  stride 144
  __shared__ __align__(32) __bf16 Vt[128 * 80];  // [hd][key],  stride 80
  __shared__ __align__(32) __bf16 Ps[64 * 80];   // [q_local][key], per-wave slabs

  for (int kt = 0; kt <= qt; kt++) {
    const int kbase = kt * 64;
    __syncthreads();
    // stage K tile row-major and V tile transposed
    for (int idx = tid; idx < 64 * 8; idx += 128) {
      const int key = idx >> 3, c16 = (idx & 7) * 16;
      const int grow = imin(kbase + key, Md - 1);
      *(v16bf*)&Ks[key * 144 + c16] =
          *(const v16bf*)&k[bOff + (long long)grow * Dc + hOff + c16];
      v16bf tmp = *(const v16bf*)&v[bOff + (long long)grow * Dc + hOff + c16];
#pragma unroll
      for (int e = 0; e < 16; e++) Vt[(c16 + e) * 80 + key] = tmp[e];
    }
    __syncthreads();

    // S = Q * K^T  (all operands resident: Q in VGPRs, K in LDS)
    v8f sacc[4] = {};
#pragma unroll
    for (int c = 0; c < 4; c++) {
      const int krow = c * 16 + l16;
#pragma unroll
      for (int kk = 0; kk < 4; kk++) {
        v16bf bf = *(const v16bf*)&Ks[krow * 144 + kk * 32 + lh * 16];
        sacc[c] = WMMA_BF16(qf[kk], bf, sacc[c]);
      }
    }

    // online softmax update (rows live in 16-lane halves; reduce via shfl_xor)
    float scl[8];
#pragma unroll
    for (int i = 0; i < 8; i++) {
      const int qrow = qbase + lh * 8 + i;
      float s0 = sacc[0][i], s1 = sacc[1][i], s2 = sacc[2][i], s3 = sacc[3][i];
      const int c0 = kbase + 0 * 16 + l16, c1 = kbase + 1 * 16 + l16;
      const int c2 = kbase + 2 * 16 + l16, c3 = kbase + 3 * 16 + l16;
      if (c0 > qrow || c0 >= Md) s0 = -1e30f;
      if (c1 > qrow || c1 >= Md) s1 = -1e30f;
      if (c2 > qrow || c2 >= Md) s2 = -1e30f;
      if (c3 > qrow || c3 >= Md) s3 = -1e30f;
      float lm = fmaxf(fmaxf(s0, s1), fmaxf(s2, s3));
      for (int mm = 1; mm < 16; mm <<= 1) lm = fmaxf(lm, __shfl_xor(lm, mm, 32));
      const float mn = fmaxf(mrow[i], lm);
      const float sc = __expf(mrow[i] - mn);
      const float p0 = __expf(s0 - mn), p1 = __expf(s1 - mn);
      const float p2 = __expf(s2 - mn), p3 = __expf(s3 - mn);
      const int prow = (wv * 16 + lh * 8 + i) * 80;
      Ps[prow + 0 * 16 + l16] = (__bf16)p0;
      Ps[prow + 1 * 16 + l16] = (__bf16)p1;
      Ps[prow + 2 * 16 + l16] = (__bf16)p2;
      Ps[prow + 3 * 16 + l16] = (__bf16)p3;
      float rs = p0 + p1 + p2 + p3;
      for (int mm = 1; mm < 16; mm <<= 1) rs += __shfl_xor(rs, mm, 32);
      lrow[i] = lrow[i] * sc + rs;
      mrow[i] = mn;
      scl[i] = sc;
    }

#pragma unroll
    for (int t = 0; t < 8; t++)
#pragma unroll
      for (int i = 0; i < 8; i++) oacc[t][i] *= scl[i];

    // O += P * V   (A = P from LDS, B = V^T-staged from LDS)
#pragma unroll
    for (int ks = 0; ks < 2; ks++) {
      v16bf af = *(const v16bf*)&Ps[(wv * 16 + l16) * 80 + ks * 32 + lh * 16];
#pragma unroll
      for (int t = 0; t < 8; t++) {
        v16bf bf = *(const v16bf*)&Vt[(t * 16 + l16) * 80 + ks * 32 + lh * 16];
        oacc[t] = WMMA_BF16(af, bf, oacc[t]);
      }
    }
  }

#pragma unroll
  for (int t = 0; t < 8; t++)
#pragma unroll
    for (int i = 0; i < 8; i++) {
      const int qrow = qbase + lh * 8 + i;
      if (qrow < Md) {
        const float o = oacc[t][i] / lrow[i];
        out[bOff + (long long)qrow * Dc + hOff + t * 16 + l16] = (__bf16)o;
      }
    }
}

// ---------------------------------------------------------------------------
// act = silu(gate) * up   (bf16, in-place into gate buffer)
// ---------------------------------------------------------------------------
__global__ void silu_mul_kernel(const __bf16* __restrict__ gate, const __bf16* __restrict__ up,
                                __bf16* __restrict__ act, const int* __restrict__ kdev) {
  const long long i = (long long)blockIdx.x * 256 + threadIdx.x;  // within batch
  const int row = (int)(i / Fc);
  if (row >= *kdev) return;
  const long long idx = (long long)blockIdx.y * Sc * (long long)Fc + i;
  const float g = (float)gate[idx];
  const float u = (float)up[idx];
  act[idx] = (__bf16)((g / (1.f + __expf(-g))) * u);
}

// ---------------------------------------------------------------------------
// Host-side orchestration
// ---------------------------------------------------------------------------
extern "C" void kernel_launch(void* const* d_in, const int* in_sizes, int n_in,
                              void* d_out, int out_size, void* d_ws, size_t ws_size,
                              hipStream_t stream) {
  const float* hidden   = (const float*)d_in[0];
  const float* vmask    = (const float*)d_in[1];
  const float* router_w = (const float*)d_in[2];
  const float* router_b = (const float*)d_in[3];
  const float* ln1_w    = (const float*)d_in[4];
  const float* ln2_w    = (const float*)d_in[5];
  const float* q_w      = (const float*)d_in[6];
  const float* k_w      = (const float*)d_in[7];
  const float* v_w      = (const float*)d_in[8];
  const float* o_w      = (const float*)d_in[9];
  const float* gate_w   = (const float*)d_in[10];
  const float* up_w     = (const float*)d_in[11];
  const float* down_w   = (const float*)d_in[12];
  float* out = (float*)d_out;

  char* ws = (char*)d_ws;
  size_t off = 0;
  auto alloc = [&](size_t bytes) -> void* {
    off = (off + 255) & ~(size_t)255;
    void* p = ws + off;
    off += bytes;
    return p;
  };
  const long long DD = (long long)Dc * Dc, FD = (long long)Fc * Dc;
  const long long SD = (long long)Sc * Dc, SF = (long long)Sc * Fc;

  __bf16* wq = (__bf16*)alloc(DD * 2);
  __bf16* wk = (__bf16*)alloc(DD * 2);
  __bf16* wvv = (__bf16*)alloc(DD * 2);
  __bf16* wo = (__bf16*)alloc(DD * 2);
  __bf16* wg = (__bf16*)alloc(FD * 2);
  __bf16* wu = (__bf16*)alloc(FD * 2);
  __bf16* wd = (__bf16*)alloc(FD * 2);
  float* route_prob = (float*)alloc((long long)Bc * Sc * 4);
  float* logp       = (float*)alloc((long long)Bc * Sc * 2 * 4);
  int*   selflag    = (int*)alloc((long long)Bc * Sc * 4);
  int*   sel_idx    = (int*)alloc((long long)Bc * Sc * 4);
  float* rwt        = (float*)alloc((long long)Bc * Sc * 4);
  int*   kdev       = (int*)alloc(256);
  float* auxpart    = (float*)alloc(256);
  __bf16* xln  = (__bf16*)alloc(Bc * SD * 2);
  __bf16* qb   = (__bf16*)alloc(Bc * SD * 2);
  __bf16* kb   = (__bf16*)alloc(Bc * SD * 2);
  __bf16* vb   = (__bf16*)alloc(Bc * SD * 2);
  __bf16* ao   = (__bf16*)alloc(Bc * SD * 2);
  float*  xmid = (float*)alloc(Bc * SD * 4);
  __bf16* hln  = (__bf16*)alloc(Bc * SD * 2);
  __bf16* gate = (__bf16*)alloc(Bc * SF * 2);
  __bf16* up   = (__bf16*)alloc(Bc * SF * 2);
  __bf16* act  = gate;  // silu*up written in place over gate

  // unselected rows of hidden_out = hidden (selected rows overwritten by down-proj scatter)
  hipMemcpyAsync(d_out, d_in[0], (size_t)Bc * SD * 4, hipMemcpyDeviceToDevice, stream);

  auto conv = [&](const float* s, __bf16* d, long long n) {
    const int blocks = (int)((n / 4 + 255) / 256);
    f32_to_bf16_kernel<<<blocks, 256, 0, stream>>>(s, d, n);
  };
  conv(q_w, wq, DD); conv(k_w, wk, DD); conv(v_w, wvv, DD); conv(o_w, wo, DD);
  conv(gate_w, wg, FD); conv(up_w, wu, FD); conv(down_w, wd, FD);

  vmask_topk_kernel<<<1, 256, 0, stream>>>(vmask, Bc * Sc, kdev);
  router_kernel<<<dim3(Sc, Bc), 256, 0, stream>>>(hidden, ln1_w, router_w, router_b,
                                                  route_prob, logp);
  rank_kernel<<<Bc, 256, 0, stream>>>(route_prob, vmask, kdev, selflag);
  compact_kernel<<<Bc, 1, 0, stream>>>(selflag, route_prob, logp, sel_idx, rwt, auxpart);
  aux_finalize_kernel<<<1, 1, 0, stream>>>(auxpart, out);

  // x_ln = rmsnorm(gather(hidden, sel_idx)) * ln1_w  -> bf16
  rmsnorm_bf16_kernel<<<dim3(Sc, Bc), 256, 0, stream>>>(hidden, sel_idx, ln1_w, xln, kdev);

  const dim3 gDD(Dc / 128, Sc / 128, Bc);
  gemm_bf16_wmma<0><<<gDD, 256, 0, stream>>>(xln, SD, wq, qb, SD, Dc, Dc, kdev,
                                             nullptr, nullptr, nullptr);
  gemm_bf16_wmma<0><<<gDD, 256, 0, stream>>>(xln, SD, wk, kb, SD, Dc, Dc, kdev,
                                             nullptr, nullptr, nullptr);
  gemm_bf16_wmma<0><<<gDD, 256, 0, stream>>>(xln, SD, wvv, vb, SD, Dc, Dc, kdev,
                                             nullptr, nullptr, nullptr);

  rope_kernel<<<dim3(Sc, Bc), 256, 0, stream>>>(qb, kb, kdev);

  flash_attn_kernel<<<dim3(Sc / 64, Hc, Bc), 128, 0, stream>>>(qb, kb, vb, ao, kdev);

  // xmid = gather(hidden) + attn_out @ o_w^T
  gemm_bf16_wmma<1><<<gDD, 256, 0, stream>>>(ao, SD, wo, xmid, SD, Dc, Dc, kdev,
                                             hidden, sel_idx, nullptr);

  // h_ln = rmsnorm(xmid) * ln2_w -> bf16
  rmsnorm_bf16_kernel<<<dim3(Sc, Bc), 256, 0, stream>>>(xmid, nullptr, ln2_w, hln, kdev);

  const dim3 gDF(Fc / 128, Sc / 128, Bc);
  gemm_bf16_wmma<0><<<gDF, 256, 0, stream>>>(hln, SD, wg, gate, SF, Fc, Dc, kdev,
                                             nullptr, nullptr, nullptr);
  gemm_bf16_wmma<0><<<gDF, 256, 0, stream>>>(hln, SD, wu, up, SF, Fc, Dc, kdev,
                                             nullptr, nullptr, nullptr);

  silu_mul_kernel<<<dim3((unsigned)(SF / 256), Bc), 256, 0, stream>>>(gate, up, act, kdev);

  // hidden_out[b, sel_idx[m], :] = xmid[m] + (act @ down_w^T) * r_weight[m]
  gemm_bf16_wmma<2><<<gDD, 256, 0, stream>>>(act, SF, wd, out, 0, Dc, Fc, kdev,
                                             xmid, sel_idx, rwt);
}
